// SelfAttention_79972291051802
// MI455X (gfx1250) — matrix operand order repaired and checked by reading, not compiled, not run on hardware
//
#include <hip/hip_runtime.h>

// ---------------------------------------------------------------------------
// Self-attention (B=8, H=W=64, C=256, C8=32) for gfx1250 (MI455X).
// bf16 WMMA everywhere, flash-attention with async global->LDS double buffer.
// ---------------------------------------------------------------------------

#define AS1 __attribute__((address_space(1)))
#define AS3 __attribute__((address_space(3)))

typedef __attribute__((ext_vector_type(16))) __bf16 v16bf;
typedef __attribute__((ext_vector_type(2)))  __bf16 v2bf;
typedef __attribute__((ext_vector_type(8)))  float  v8f;
typedef int v4i __attribute__((vector_size(16)));   // matches async builtin param

union Frag16 {
    unsigned int u[8];
    uint4        q[2];
    v16bf        v;
};

// ---- f32 -> bf16 conversion ----
// Prefer v_cvt_pk_bf16_f32 if the toolchain has it; otherwise round-half-up
// (+0x8000) and splice the two high halves with a single v_perm_b32.
#ifndef ATHENA_NO_CVT_PK
#if __has_builtin(__builtin_amdgcn_cvt_pk_bf16_f32)
#define ATHENA_CVT_PK 1
#endif
#endif

__device__ __forceinline__ unsigned int pack_bf16(float lo, float hi) {
#ifdef ATHENA_CVT_PK
    union { v2bf v; unsigned int u; } c;
    c.v = __builtin_amdgcn_cvt_pk_bf16_f32(lo, hi);
    return c.u;
#else
    return __builtin_amdgcn_perm(__float_as_uint(hi) + 0x8000u,
                                 __float_as_uint(lo) + 0x8000u,
                                 0x07060302u);      // {hi[31:16], lo[31:16]}
#endif
}

__device__ __forceinline__ unsigned short f32_to_bf16(float f) {
    unsigned int u = __float_as_uint(f) + 0x8000u;
    return (unsigned short)(u >> 16);               // selects *_d16_hi_b16 stores
}

// ---- async global -> LDS (CDNA5 GLOBAL_LOAD_ASYNC_TO_LDS_B128) ----
#if __has_builtin(__builtin_amdgcn_global_load_async_to_lds_b128)
#define ATHENA_ASYNC 1
#endif

__device__ __forceinline__ void cp16_async(const unsigned int* g, unsigned int* l) {
#ifdef ATHENA_ASYNC
    __builtin_amdgcn_global_load_async_to_lds_b128(
        (AS1 v4i*)g, (AS3 v4i*)l, 0, 0);
#else
    *(uint4*)l = *(const uint4*)g;          // synchronous fallback
#endif
}

__device__ __forceinline__ void wait_async0() {
#ifdef ATHENA_ASYNC
#if __has_builtin(__builtin_amdgcn_s_wait_asynccnt)
    __builtin_amdgcn_s_wait_asynccnt(0);
#else
    asm volatile("s_wait_asynccnt 0x0" ::: "memory");
#endif
#endif
}

// A-fragment (16x32 bf16): lane row = lane%16; out_proj A and B: VGPR v, group g hold K pair at
//   ((v&4)?16:0) + g*8 + (v&3)*2   -> dwords [row*16+g*4 .. +3] and [+8 .. +11]
// B-fragment (32x16 bf16): lane col = lane%16; qkv_proj/flash_attn A and B: VGPR v, group g: K pair g*16+2v
//   -> dwords [col*16+g*8 .. +7]
__device__ __forceinline__ v8f wmma_bf16(const Frag16& a, const Frag16& b, v8f c) {
    return __builtin_amdgcn_wmma_f32_16x16x32_bf16(
        false, a.v, false, b.v, (short)0, c, false, false);
}

#define LOG2E 1.4426950408889634f

// ---------------------------------------------------------------------------
// Kernel 0: weight prep. WqT/WkT/WvT: [32][256] bf16 transposed; WoT: [256][32].
// Wq is pre-scaled by log2(e) so softmax runs in exp2 domain.
// ---------------------------------------------------------------------------
__global__ __launch_bounds__(256) void prep_weights(
    const float* __restrict__ Wq, const float* __restrict__ Wk,
    const float* __restrict__ Wv, const float* __restrict__ Wo,
    unsigned short* __restrict__ wqT, unsigned short* __restrict__ wkT,
    unsigned short* __restrict__ wvT, unsigned short* __restrict__ woT)
{
    int i = blockIdx.x * 256 + threadIdx.x;          // 8192 = 256*32
    if (i < 256 * 32) {
        int c = i >> 5;           // 0..255
        int d = i & 31;           // 0..31
        wqT[d * 256 + c] = f32_to_bf16(Wq[c * 32 + d] * LOG2E);
        wkT[d * 256 + c] = f32_to_bf16(Wk[c * 32 + d]);
        wvT[d * 256 + c] = f32_to_bf16(Wv[c * 32 + d]);
        woT[c * 32 + d]  = f32_to_bf16(Wo[d * 256 + c]);
    }
}

// ---------------------------------------------------------------------------
// Kernel 1: QKV projection + ReLU.  q,k row-major bf16 [B*N][32];
// v transposed bf16 per batch: vT[(b*32+d)*4096 + pix].  q in log2 domain.
// ---------------------------------------------------------------------------
__global__ __launch_bounds__(256) void qkv_proj(
    const float* __restrict__ x,
    const unsigned short* __restrict__ wqT,
    const unsigned short* __restrict__ wkT,
    const unsigned short* __restrict__ wvT,
    const float* __restrict__ bq, const float* __restrict__ bk,
    const float* __restrict__ bv,
    unsigned short* __restrict__ qo, unsigned short* __restrict__ ko,
    unsigned short* __restrict__ vTo)
{
    const int lane = threadIdx.x & 31;
    const int wave = threadIdx.x >> 5;
    const int tile = blockIdx.x * 8 + wave;        // 2048 tiles of 16 pixels
    const int pix0 = tile * 16;
    const int m = lane & 15;
    const int g = lane >> 4;

    const float* xrow = x + (pix0 + m) * 256 + g * 16;  // A-frag source row

    const unsigned int* wq32 = (const unsigned int*)wqT;
    const unsigned int* wk32 = (const unsigned int*)wkT;
    const unsigned int* wv32 = (const unsigned int*)wvT;
    const int wbase0 = m * 128 + g * 8;             // t=0 column
    const int wbase1 = wbase0 + 16 * 128;           // t=1 column

    v8f accq[2] = {{}, {}}, acck[2] = {{}, {}}, accv[2] = {{}, {}};

    for (int kc = 0; kc < 8; ++kc) {
        Frag16 a;
        {
            float4 f0 = *(const float4*)(xrow + kc * 32);
            float4 f1 = *(const float4*)(xrow + kc * 32 + 4);
            float4 f2 = *(const float4*)(xrow + kc * 32 + 8);
            float4 f3 = *(const float4*)(xrow + kc * 32 + 12);
            a.u[0] = pack_bf16(f0.x, f0.y); a.u[1] = pack_bf16(f0.z, f0.w);
            a.u[2] = pack_bf16(f1.x, f1.y); a.u[3] = pack_bf16(f1.z, f1.w);
            a.u[4] = pack_bf16(f2.x, f2.y); a.u[5] = pack_bf16(f2.z, f2.w);
            a.u[6] = pack_bf16(f3.x, f3.y); a.u[7] = pack_bf16(f3.z, f3.w);
        }
        #pragma unroll
        for (int t = 0; t < 2; ++t) {
            const int idx = (t ? wbase1 : wbase0) + kc * 16;
            Frag16 fq, fk, fv;
            fq.q[0] = *(const uint4*)&wq32[idx]; fq.q[1] = *(const uint4*)&wq32[idx + 4];
            fk.q[0] = *(const uint4*)&wk32[idx]; fk.q[1] = *(const uint4*)&wk32[idx + 4];
            fv.q[0] = *(const uint4*)&wv32[idx]; fv.q[1] = *(const uint4*)&wv32[idx + 4];
            accq[t] = wmma_bf16(a, fq, accq[t]);
            acck[t] = wmma_bf16(a, fk, acck[t]);
            accv[t] = wmma_bf16(a, fv, accv[t]);
        }
    }

    // epilogue: bias + ReLU, D layout: row = v + 8*g, col = lane%16 (+16*t)
    #pragma unroll
    for (int t = 0; t < 2; ++t) {
        int n = (lane & 15) + 16 * t;
        float biasq = bq[n] * LOG2E, biask = bk[n], biasv = bv[n];
        #pragma unroll
        for (int v = 0; v < 8; ++v) {
            int dm  = v + 8 * g;
            int pix = pix0 + dm;
            float rq = accq[t][v] + biasq; rq = rq > 0.f ? rq : 0.f;
            float rk = acck[t][v] + biask; rk = rk > 0.f ? rk : 0.f;
            float rv = accv[t][v] + biasv; rv = rv > 0.f ? rv : 0.f;
            qo[pix * 32 + n] = f32_to_bf16(rq);
            ko[pix * 32 + n] = f32_to_bf16(rk);
            int b = pix >> 12, w = pix & 4095;
            vTo[(b * 32 + n) * 4096 + w] = f32_to_bf16(rv);
        }
    }
}

// ---------------------------------------------------------------------------
// Kernel 2: flash attention, async double-buffered K/V staging.
// 8 waves/WG, each wave owns 16 query rows, 128 chunks of 32 keys.
// Softmax in exp2 domain; group-max stabilization; per-lane partial row sums.
// ---------------------------------------------------------------------------
__global__ __launch_bounds__(256) void flash_attn(
    const unsigned short* __restrict__ q,
    const unsigned short* __restrict__ k,
    const unsigned short* __restrict__ vT,
    float* __restrict__ O)
{
    __shared__ unsigned int   ks[2][512];          // 32 keys x 32 dims bf16
    __shared__ unsigned int   vs[2][512];          // 32 dims x 32 keys bf16 (vT)
    __shared__ unsigned short ps[8][16 * 32];      // per-wave P tile, 16x32 bf16

    const int tid  = threadIdx.x;
    const int lane = tid & 31;
    const int wave = tid >> 5;
    const int b    = blockIdx.x >> 5;                      // 8 batches
    const int q0   = (blockIdx.x & 31) * 128 + wave * 16;  // query row base
    const int m = lane & 15;
    const int g = lane >> 4;

    const unsigned int* q32 = (const unsigned int*)q;
    const unsigned int* k32 = (const unsigned int*)k;
    const unsigned int* v32 = (const unsigned int*)vT;

    // chunk-invariant staging addresses: 256 threads x 16B = 4KB (ks + vs)
    const unsigned int* gK = &k32[(b * 4096 + (tid >> 2)) * 16 + ((tid & 3) << 2)];
    const unsigned int* gV = &v32[(b * 32 + ((tid - 128) >> 2)) * 2048 + (((tid - 128) & 3) << 2)];

    auto issue = [&](int buf, int kc) {
        if (tid < 128) cp16_async(gK + kc * 32 * 16, &ks[buf][tid << 2]);
        else           cp16_async(gV + kc * 16,      &vs[buf][(tid - 128) << 2]);
    };

    // query A-fragment (row-major bf16; two contiguous 16B groups per lane)
    Frag16 aq;
    {
        const unsigned int* qb = q32 + (b * 4096 + q0 + m) * 16 + g * 8;
        aq.q[0] = *(const uint4*)(qb);
        aq.q[1] = *(const uint4*)(qb + 4);
    }

    const int brow0 = (lane & 15) * 16 + g * 8;    // B-frag dword base, cols 0..15
    const int brow1 = brow0 + 256;                 // cols 16..31

    v8f o0 = {}, o1 = {};
    float psum[8];
    #pragma unroll
    for (int v = 0; v < 8; ++v) psum[v] = 0.f;
    float gmax = -1e30f;

    issue(0, 0);

    #pragma unroll 2
    for (int kc = 0; kc < 128; ++kc) {
        const int buf = kc & 1;
        wait_async0();                  // own chunk data landed
        __syncthreads();                // everyone's landed; prev buffer free
        if (kc + 1 < 128) issue(buf ^ 1, kc + 1);

        // ---- S = q k^T (log2 domain) : two 16x16 tiles ----
        Frag16 bk0, bk1;
        bk0.q[0] = *(const uint4*)&ks[buf][brow0];
        bk0.q[1] = *(const uint4*)&ks[buf][brow0 + 4];
        bk1.q[0] = *(const uint4*)&ks[buf][brow1];
        bk1.q[1] = *(const uint4*)&ks[buf][brow1 + 4];
        v8f z = {};
        v8f s0 = wmma_bf16(aq, bk0, z);
        v8f s1 = wmma_bf16(aq, bk1, z);

        // ---- online softmax: one shared max per 16-lane group ----
        float mx = fmaxf(s0[0], s1[0]);
        #pragma unroll
        for (int v = 1; v < 8; ++v) mx = fmaxf(mx, fmaxf(s0[v], s1[v]));
        mx = fmaxf(mx, __shfl_xor(mx, 1, 16));
        mx = fmaxf(mx, __shfl_xor(mx, 2, 16));
        mx = fmaxf(mx, __shfl_xor(mx, 4, 16));
        mx = fmaxf(mx, __shfl_xor(mx, 8, 16));
        float nm    = fmaxf(gmax, mx);
        float scale = exp2f(gmax - nm);
        gmax = nm;

        float p0[8], p1[8];
        #pragma unroll
        for (int v = 0; v < 8; ++v) {
            float e0 = exp2f(s0[v] - nm);
            float e1 = exp2f(s1[v] - nm);
            psum[v] = psum[v] * scale + (e0 + e1);   // per-lane partial row sum
            o0[v] *= scale;
            o1[v] *= scale;
            p0[v] = e0;
            p1[v] = e1;
        }

        // ---- restage P (D layout) -> A layout via per-wave LDS ----
        unsigned short* pw = ps[wave];
        #pragma unroll
        for (int v = 0; v < 8; ++v) {
            int pm = v + 8 * g;
            pw[pm * 32 + (lane & 15)]      = f32_to_bf16(p0[v]);
            pw[pm * 32 + (lane & 15) + 16] = f32_to_bf16(p1[v]);
        }
        asm volatile("s_wait_dscnt 0x0" ::: "memory");
        __builtin_amdgcn_wave_barrier();

        Frag16 ap;
        {
            const unsigned int* pw32 = (const unsigned int*)pw + m * 16 + g * 8;
            ap.q[0] = *(const uint4*)(pw32);
            ap.q[1] = *(const uint4*)(pw32 + 4);
        }

        // ---- O += P V : B-frag from vT chunk (K = key, pairs contiguous) ----
        Frag16 bv0, bv1;
        bv0.q[0] = *(const uint4*)&vs[buf][brow0];
        bv0.q[1] = *(const uint4*)&vs[buf][brow0 + 4];
        bv1.q[0] = *(const uint4*)&vs[buf][brow1];
        bv1.q[1] = *(const uint4*)&vs[buf][brow1 + 4];
        o0 = wmma_bf16(ap, bv0, o0);
        o1 = wmma_bf16(ap, bv1, o1);
    }

    // ---- final row-sum reduction, normalize, store ----
    #pragma unroll
    for (int v = 0; v < 8; ++v) {
        float s = psum[v];
        s += __shfl_xor(s, 1, 16);
        s += __shfl_xor(s, 2, 16);
        s += __shfl_xor(s, 4, 16);
        s += __shfl_xor(s, 8, 16);
        float inv = 1.f / s;
        int dm = v + 8 * g;
        int base = (b * 4096 + q0 + dm) * 32 + (lane & 15);
        O[base]      = o0[v] * inv;
        O[base + 16] = o1[v] * inv;
    }
}

// ---------------------------------------------------------------------------
// Kernel 3: out = ReLU(O @ Wo + bo), K=32, one wave per 16-pixel tile,
// 16 column tiles of 16 channels.
// ---------------------------------------------------------------------------
__global__ __launch_bounds__(256) void out_proj(
    const float* __restrict__ O,
    const unsigned short* __restrict__ woT,     // [256][32] bf16
    const float* __restrict__ bo,
    float* __restrict__ out)
{
    const int lane = threadIdx.x & 31;
    const int wave = threadIdx.x >> 5;
    const int tile = blockIdx.x * 8 + wave;
    const int pix0 = tile * 16;
    const int m = lane & 15;
    const int g = lane >> 4;

    Frag16 a;
    {
        const float* ob = O + (pix0 + m) * 32 + g * 8;
        float4 f0 = *(const float4*)(ob);
        float4 f1 = *(const float4*)(ob + 4);
        float4 f2 = *(const float4*)(ob + 16);
        float4 f3 = *(const float4*)(ob + 20);
        a.u[0] = pack_bf16(f0.x, f0.y); a.u[1] = pack_bf16(f0.z, f0.w);
        a.u[2] = pack_bf16(f1.x, f1.y); a.u[3] = pack_bf16(f1.z, f1.w);
        a.u[4] = pack_bf16(f2.x, f2.y); a.u[5] = pack_bf16(f2.z, f2.w);
        a.u[6] = pack_bf16(f3.x, f3.y); a.u[7] = pack_bf16(f3.z, f3.w);
    }

    const unsigned int* w32 = (const unsigned int*)woT;
    for (int ct = 0; ct < 16; ++ct) {
        int n = ct * 16 + (lane & 15);          // output channel
        Frag16 bf;
        bf.q[0] = *(const uint4*)&w32[n * 16 + g * 4];
        bf.q[1] = *(const uint4*)&w32[n * 16 + g * 4 + 8];
        v8f acc = {};
        acc = wmma_bf16(a, bf, acc);
        float bias = bo[n];
        #pragma unroll
        for (int v = 0; v < 8; ++v) {
            int dm = v + 8 * g;
            float r = acc[v] + bias;
            out[(pix0 + dm) * 256 + n] = r > 0.f ? r : 0.f;
        }
    }
}

// ---------------------------------------------------------------------------
// Workspace layout (bytes):
//   0        : WqT bf16 [32][256]   (16 KB)   -- pre-scaled by log2(e)
//   16K      : WkT bf16 [32][256]   (16 KB)
//   32K      : WvT bf16 [32][256]   (16 KB)
//   48K      : WoT bf16 [256][32]   (16 KB)
//   64K      : q  bf16 [8*4096][32]  (2 MB)
//   64K+2M   : k  bf16 [8*4096][32]  (2 MB)
//   64K+4M   : vT bf16 [8][32][4096] (2 MB)
//   64K+6M   : O  f32  [8*4096][32]  (4 MB)
// ---------------------------------------------------------------------------
extern "C" void kernel_launch(void* const* d_in, const int* in_sizes, int n_in,
                              void* d_out, int out_size, void* d_ws, size_t ws_size,
                              hipStream_t stream) {
    const float* x  = (const float*)d_in[0];
    const float* Wq = (const float*)d_in[1];
    const float* bq = (const float*)d_in[2];
    const float* Wk = (const float*)d_in[3];
    const float* bk = (const float*)d_in[4];
    const float* Wv = (const float*)d_in[5];
    const float* bv = (const float*)d_in[6];
    const float* Wo = (const float*)d_in[7];
    const float* bo = (const float*)d_in[8];
    float* out = (float*)d_out;

    char* ws = (char*)d_ws;
    const size_t KB = 1024, MB = 1024 * 1024;
    unsigned short* wqT = (unsigned short*)(ws);
    unsigned short* wkT = (unsigned short*)(ws + 16 * KB);
    unsigned short* wvT = (unsigned short*)(ws + 32 * KB);
    unsigned short* woT = (unsigned short*)(ws + 48 * KB);
    unsigned short* qb  = (unsigned short*)(ws + 64 * KB);
    unsigned short* kb  = (unsigned short*)(ws + 64 * KB + 2 * MB);
    unsigned short* vTb = (unsigned short*)(ws + 64 * KB + 4 * MB);
    float*          Ob  = (float*)        (ws + 64 * KB + 6 * MB);

    prep_weights<<<32, 256, 0, stream>>>(Wq, Wk, Wv, Wo, wqT, wkT, wvT, woT);
    qkv_proj<<<256, 256, 0, stream>>>(x, wqT, wkT, wvT, bq, bk, bv, qb, kb, vTb);
    flash_attn<<<256, 256, 0, stream>>>(qb, kb, vTb, Ob);
    out_proj<<<256, 256, 0, stream>>>(Ob, woT, bo, out);
}
